// FusedMoE_3685081940534
// MI455X (gfx1250) — matrix operand order repaired and checked by reading, not compiled, not run on hardware
//
#include <hip/hip_runtime.h>
#include <hip/hip_bf16.h>

// Problem constants (from reference)
#define T_TOK 2048
#define H_DIM 1024
#define F_DIM 2816
#define E_EXP 8
#define K_TOP 2
#define TK (T_TOK * K_TOP)
#define MT 64              // rows per block tile (4 WMMA M-subtiles)

typedef __attribute__((ext_vector_type(16))) __bf16 v16bf;
typedef __attribute__((ext_vector_type(8)))  float  v8f;

// ---------------------------------------------------------------------------
// helpers
// ---------------------------------------------------------------------------
union BfPack {
    v16bf v;
    __bf16 h[16];
    uint4 q[2];
};

// Convert 16 f32 -> 16 bf16 using native casts (lowers to v_cvt_pk_bf16_f32).
__device__ __forceinline__ BfPack cvt16(float4 f0, float4 f1, float4 f2, float4 f3) {
    BfPack r;
    r.h[0]  = (__bf16)f0.x; r.h[1]  = (__bf16)f0.y; r.h[2]  = (__bf16)f0.z; r.h[3]  = (__bf16)f0.w;
    r.h[4]  = (__bf16)f1.x; r.h[5]  = (__bf16)f1.y; r.h[6]  = (__bf16)f1.z; r.h[7]  = (__bf16)f1.w;
    r.h[8]  = (__bf16)f2.x; r.h[9]  = (__bf16)f2.y; r.h[10] = (__bf16)f2.z; r.h[11] = (__bf16)f2.w;
    r.h[12] = (__bf16)f3.x; r.h[13] = (__bf16)f3.y; r.h[14] = (__bf16)f3.z; r.h[15] = (__bf16)f3.w;
    return r;
}

// Fast reciprocal (v_rcp_f32) for the SiLU epilogue; avoids the precise
// IEEE division expansion (~10 VALU ops/elem).
__device__ __forceinline__ float fast_rcp(float x) {
#if __has_builtin(__builtin_amdgcn_rcpf)
    return __builtin_amdgcn_rcpf(x);
#else
    return 1.0f / x;
#endif
}

__device__ __forceinline__ float silu(float g) {
    return g * fast_rcp(1.0f + __expf(-g));
}

// B-operand (32x16 bf16): lanes 0..15 = col n hold K-octets {0..7,16..23},
// lanes 16..31 hold {8..15,24..31}; weight rows are K-contiguous in memory.
__device__ __forceinline__ v16bf load_b_f32(const float* __restrict__ row,
                                            int kbase, int khalf) {
    const float* p = row + kbase + khalf * 8;
    float4 f0 = *reinterpret_cast<const float4*>(p);
    float4 f1 = *reinterpret_cast<const float4*>(p + 4);
    float4 f2 = *reinterpret_cast<const float4*>(p + 16);
    float4 f3 = *reinterpret_cast<const float4*>(p + 20);
    return cvt16(f0, f1, f2, f3).v;
}

// A-operand (16x32 bf16): lane<16 -> K-octets {0..7,16..23}, lane>=16 ->
// {8..15,24..31}; memory dword order matches fragment order -> 2 x b128.
__device__ __forceinline__ v16bf load_a_bf16(const unsigned short* __restrict__ row,
                                             int half) {
    BfPack a;
    a.q[0] = *reinterpret_cast<const uint4*>(row + half * 8);
    a.q[1] = *reinterpret_cast<const uint4*>(row + 16 + half * 8);
    return a.v;
}

__device__ __forceinline__ v8f wmma_bf16(v16bf a, v16bf b, v8f c) {
    return __builtin_amdgcn_wmma_f32_16x16x32_bf16(
        /*neg_a=*/false, a, /*neg_b=*/false, b,
        /*c_mod=*/(short)0, c, /*reuse_a=*/false, /*reuse_b=*/false);
}

// ---------------------------------------------------------------------------
// routing kernels
// ---------------------------------------------------------------------------
__global__ void moe_zero_counts(int* counts, int* fillc) {
    int i = threadIdx.x;
    if (i < E_EXP) { counts[i] = 0; fillc[i] = 0; }
}

__global__ void moe_route_count(const int* __restrict__ ids, int* counts) {
    int i = blockIdx.x * blockDim.x + threadIdx.x;
    if (i < TK) atomicAdd(&counts[ids[i] & (E_EXP - 1)], 1);
}

__global__ void moe_scan(const int* __restrict__ counts, int* offsets) {
    if (threadIdx.x == 0) {
        int s = 0;
        for (int e = 0; e < E_EXP; ++e) { offsets[e] = s; s += counts[e]; }
        offsets[E_EXP] = s;
    }
}

__global__ void moe_fill(const int* __restrict__ ids, int* fillc,
                         const int* __restrict__ offsets,
                         int* row_tok, int* arow_map) {
    int i = blockIdx.x * blockDim.x + threadIdx.x;
    if (i < TK) {
        int e = ids[i] & (E_EXP - 1);
        int pos = atomicAdd(&fillc[e], 1);
        int r = offsets[e] + pos;
        row_tok[r] = i / K_TOP;
        arow_map[i] = r;
    }
}

// ---------------------------------------------------------------------------
// gate_up GEMM + SiLU*up -> bf16 act
// block tile: 64 rows x 64 f-cols (gate AND up). 128 threads = 4 waves,
// wave w owns 16 cols. A chunk (64x32) staged in LDS as bf16 once per k-step
// and shared by all 4 waves (8 WMMAs per staged chunk per wave).
// grid: (F/64, TK/64, E)
// ---------------------------------------------------------------------------
__global__ __launch_bounds__(128) void moe_gateup_silu(
    const float* __restrict__ x, const float* __restrict__ wgu,
    const int* __restrict__ counts, const int* __restrict__ offsets,
    const int* __restrict__ row_tok, unsigned short* __restrict__ act)
{
    __shared__ __align__(16) unsigned short ldsA[MT * 32];

    const int e = blockIdx.z;
    const int nrows = counts[e];
    const int mt = blockIdx.y;
    if (mt * MT >= nrows) return;
    const int seg = offsets[e];
    const int fb  = blockIdx.x * 64;

    const int tid  = threadIdx.x;
    const int lane = tid & 31;
    const int w    = tid >> 5;
    const int half = lane >> 4;
    const int nl   = lane & 15;

    // cooperative A staging assignment: thread -> (row, k-half)
    const int crow = tid >> 1;            // 0..63
    const int ck   = (tid & 1) * 16;      // 0 or 16
    int rload = mt * MT + crow;
    if (rload >= nrows) rload = nrows - 1;
    const float* xrow = x + (size_t)row_tok[seg + rload] * H_DIM;

    const float* wbase = wgu + (size_t)e * (2 * F_DIM) * H_DIM;
    const int fc = fb + w * 16 + nl;
    const float* bg = wbase + (size_t)fc * H_DIM;
    const float* bu = wbase + (size_t)(F_DIM + fc) * H_DIM;

    v8f zero = {};
    v8f accg[4], accu[4];
#pragma unroll
    for (int s = 0; s < 4; ++s) { accg[s] = zero; accu[s] = zero; }

    for (int k = 0; k < H_DIM; k += 32) {
        // stage A chunk (64 rows x 32 k) into LDS as bf16
        {
            const float* p = xrow + k + ck;
            float4 f0 = *reinterpret_cast<const float4*>(p);
            float4 f1 = *reinterpret_cast<const float4*>(p + 4);
            float4 f2 = *reinterpret_cast<const float4*>(p + 8);
            float4 f3 = *reinterpret_cast<const float4*>(p + 12);
            BfPack pk = cvt16(f0, f1, f2, f3);
            uint4* dst = reinterpret_cast<uint4*>(&ldsA[crow * 32 + ck]);
            dst[0] = pk.q[0];
            dst[1] = pk.q[1];
        }
        __syncthreads();

        __builtin_prefetch(bg + k + 64, 0, 1);
        __builtin_prefetch(bu + k + 64, 0, 1);

        v16bf afr[4];
#pragma unroll
        for (int s = 0; s < 4; ++s)
            afr[s] = load_a_bf16(&ldsA[(s * 16 + nl) * 32], half);

        v16bf b = load_b_f32(bg, k, half);
#pragma unroll
        for (int s = 0; s < 4; ++s) accg[s] = wmma_bf16(afr[s], b, accg[s]);

        b = load_b_f32(bu, k, half);
#pragma unroll
        for (int s = 0; s < 4; ++s) accu[s] = wmma_bf16(afr[s], b, accu[s]);

        __syncthreads();
    }

    // C/D layout: lane&15 = col, VGPR v = row (half*8 + v). Fuse SiLU*up.
    __bf16* actb = reinterpret_cast<__bf16*>(act);
    const int mbase = half * 8;
    const int col = fb + w * 16 + nl;
#pragma unroll
    for (int s = 0; s < 4; ++s) {
        v8f g = accg[s];
        v8f u = accu[s];
#pragma unroll
        for (int v = 0; v < 8; ++v) {
            int r = mt * MT + s * 16 + mbase + v;
            if (r < nrows) {
                float av = silu(g[v]) * u[v];
                actb[(size_t)(seg + r) * F_DIM + col] = (__bf16)av;
            }
        }
    }
}

// ---------------------------------------------------------------------------
// down GEMM: ybuf[r][h] = act[r][:] . Wd[e][h][:]
// block tile: 64 rows x 128 h-cols; wave w owns 32 cols (2 B frags),
// 4 M-subtiles share each B fragment. A is bf16 in memory already in
// fragment dword order -> raw b128 loads.
// grid: (H/128, TK/64, E)
// ---------------------------------------------------------------------------
__global__ __launch_bounds__(128) void moe_down(
    const unsigned short* __restrict__ act, const float* __restrict__ wd,
    const int* __restrict__ counts, const int* __restrict__ offsets,
    float* __restrict__ ybuf)
{
    const int e = blockIdx.z;
    const int nrows = counts[e];
    const int mt = blockIdx.y;
    if (mt * MT >= nrows) return;
    const int seg = offsets[e];
    const int hb  = blockIdx.x * 128;

    const int lane = threadIdx.x & 31;
    const int w    = threadIdx.x >> 5;
    const int half = lane >> 4;
    const int nl   = lane & 15;

    const unsigned short* arow[4];
#pragma unroll
    for (int s = 0; s < 4; ++s) {
        int r = mt * MT + s * 16 + nl;
        if (r >= nrows) r = nrows - 1;
        arow[s] = act + (size_t)(seg + r) * F_DIM;
    }

    const float* wbase = wd + (size_t)e * H_DIM * F_DIM;
    const int hc = hb + w * 32 + nl;
    const float* b0 = wbase + (size_t)(hc)      * F_DIM;
    const float* b1 = wbase + (size_t)(hc + 16) * F_DIM;

    v8f zero = {};
    v8f c0[4], c1[4];
#pragma unroll
    for (int s = 0; s < 4; ++s) { c0[s] = zero; c1[s] = zero; }

    for (int k = 0; k < F_DIM; k += 32) {
        __builtin_prefetch(b0 + k + 64, 0, 1);
        __builtin_prefetch(b1 + k + 64, 0, 1);

        v16bf afr[4];
#pragma unroll
        for (int s = 0; s < 4; ++s) afr[s] = load_a_bf16(arow[s] + k, half);

        v16bf b = load_b_f32(b0, k, half);
#pragma unroll
        for (int s = 0; s < 4; ++s) c0[s] = wmma_bf16(afr[s], b, c0[s]);

        b = load_b_f32(b1, k, half);
#pragma unroll
        for (int s = 0; s < 4; ++s) c1[s] = wmma_bf16(afr[s], b, c1[s]);
    }

    const int mbase = half * 8;
#pragma unroll
    for (int s = 0; s < 4; ++s) {
#pragma unroll
        for (int j = 0; j < 2; ++j) {
            v8f c = j ? c1[s] : c0[s];
            const int col = hb + w * 32 + j * 16 + nl;
#pragma unroll
            for (int v = 0; v < 8; ++v) {
                int r = mt * MT + s * 16 + mbase + v;
                if (r < nrows) ybuf[(size_t)(seg + r) * H_DIM + col] = c[v];
            }
        }
    }
}

// ---------------------------------------------------------------------------
// deterministic combine: out[t][h] = sum_k tw[t][k] * ybuf[arow[t][k]][h]
// ---------------------------------------------------------------------------
__global__ void moe_combine(const float* __restrict__ ybuf,
                            const float* __restrict__ tw,
                            const int* __restrict__ arow_map,
                            float* __restrict__ out)
{
    int idx = blockIdx.x * blockDim.x + threadIdx.x;
    if (idx >= T_TOK * H_DIM) return;
    int t = idx >> 10;          // H_DIM == 1024
    int h = idx & 1023;
    float s = 0.0f;
#pragma unroll
    for (int k = 0; k < K_TOP; ++k) {
        int r = arow_map[t * K_TOP + k];
        s += tw[t * K_TOP + k] * ybuf[(size_t)r * H_DIM + h];
    }
    out[idx] = s;
}

// ---------------------------------------------------------------------------
// launch
// ---------------------------------------------------------------------------
extern "C" void kernel_launch(void* const* d_in, const int* in_sizes, int n_in,
                              void* d_out, int out_size, void* d_ws, size_t ws_size,
                              hipStream_t stream) {
    const float* x   = (const float*)d_in[0];   // (T,H)
    const float* tw  = (const float*)d_in[1];   // (T,K)
    const int*   ids = (const int*)d_in[2];     // (T,K)
    const float* wgu = (const float*)d_in[3];   // (E,2F,H)
    const float* wd  = (const float*)d_in[4];   // (E,H,F)
    float* out = (float*)d_out;                 // (T,H)

    // workspace partition (~40 MB): ints | act bf16 (TK*F) | ybuf f32 (TK*H)
    char* ws = (char*)d_ws;
    int* counts   = (int*)ws;                 ws += E_EXP * sizeof(int);
    int* offsets  = (int*)ws;                 ws += (E_EXP + 1) * sizeof(int);
    int* fillc    = (int*)ws;                 ws += E_EXP * sizeof(int);
    int* row_tok  = (int*)ws;                 ws += TK * sizeof(int);
    int* arow_map = (int*)ws;                 ws += TK * sizeof(int);
    uintptr_t p = ((uintptr_t)ws + 255) & ~(uintptr_t)255;
    unsigned short* act = (unsigned short*)p; p += (size_t)TK * F_DIM * sizeof(unsigned short);
    p = (p + 255) & ~(uintptr_t)255;
    float* ybuf = (float*)p;

    moe_zero_counts<<<1, 32, 0, stream>>>(counts, fillc);
    moe_route_count<<<(TK + 255) / 256, 256, 0, stream>>>(ids, counts);
    moe_scan<<<1, 32, 0, stream>>>(counts, offsets);
    moe_fill<<<(TK + 255) / 256, 256, 0, stream>>>(ids, fillc, offsets, row_tok, arow_map);

    dim3 g1(F_DIM / 64, TK / MT, E_EXP);    // 44 x 64 x 8
    moe_gateup_silu<<<g1, 128, 0, stream>>>(x, wgu, counts, offsets, row_tok, act);

    dim3 g2(H_DIM / 128, TK / MT, E_EXP);   // 8 x 64 x 8
    moe_down<<<g2, 128, 0, stream>>>(act, wd, counts, offsets, ybuf);

    moe_combine<<<(T_TOK * H_DIM + 255) / 256, 256, 0, stream>>>(ybuf, tw, arow_map, out);
}
